// AutoEncoderTopK_22978075033978
// MI455X (gfx1250) — matrix operand order repaired and checked by reading, not compiled, not run on hardware
//
#include <hip/hip_runtime.h>

typedef __attribute__((ext_vector_type(16))) __bf16 v16bf;
typedef __attribute__((ext_vector_type(8)))  __bf16 v8bf;
typedef __attribute__((ext_vector_type(8)))  float  v8f;

#define B_SZ  8192
#define D_SZ  2048
#define F_SZ  32768
#define K_TOP 64

#define ENC_ROWS 32         // M-tile rows per block
#define ENC_THREADS 512     // 16 waves
#define ACT_PITCH 264       // padded acts row pitch (floats)

__device__ __forceinline__ unsigned short f32_to_bf16_bits(float f) {
  unsigned u = __float_as_uint(f);
  unsigned r = u + 0x7FFFu + ((u >> 16) & 1u);   // round-to-nearest-even
  return (unsigned short)(r >> 16);
}
__device__ __forceinline__ float bf16_bits_to_f32(unsigned short s) {
  return __uint_as_float(((unsigned)s) << 16);
}
__device__ __forceinline__ float bf16_to_f32(__bf16 h) {
  union { __bf16 h; unsigned short s; } cv; cv.h = h;
  return bf16_bits_to_f32(cv.s);
}

// ---------------------------------------------------------------------------
// Kernel 1: xm = bf16(x - b_dec), row-major (B, D)
// ---------------------------------------------------------------------------
__global__ __launch_bounds__(256)
void k_prep_xm(const float* __restrict__ x, const float* __restrict__ b_dec,
               unsigned short* __restrict__ xm) {
  const int row = blockIdx.x;
  const size_t base = (size_t)row * D_SZ;
  for (int d = threadIdx.x; d < D_SZ; d += 256) {
    xm[base + d] = f32_to_bf16_bits(x[base + d] - b_dec[d]);
  }
}

// ---------------------------------------------------------------------------
// Kernel 2: bulk f32 -> bf16 conversion (used for W_enc), vectorized x4
// ---------------------------------------------------------------------------
__global__ __launch_bounds__(256)
void k_f32_to_bf16_vec4(const float* __restrict__ src,
                        unsigned short* __restrict__ dst, size_t n4) {
  size_t i = (size_t)blockIdx.x * 256 + threadIdx.x;
  if (i >= n4) return;
  const float4 f = ((const float4*)src)[i];
  ushort4 o;
  o.x = f32_to_bf16_bits(f.x); o.y = f32_to_bf16_bits(f.y);
  o.z = f32_to_bf16_bits(f.z); o.w = f32_to_bf16_bits(f.w);
  ((ushort4*)dst)[i] = o;
}

// ---------------------------------------------------------------------------
// Kernel 3: W_dec (D, F) f32 -> W_decT (F, D) bf16, tiled LDS transpose
// ---------------------------------------------------------------------------
__global__ __launch_bounds__(256)
void k_transpose_wdec(const float* __restrict__ wdec,
                      unsigned short* __restrict__ wdecT) {
  __shared__ float tile[32][33];
  const int tx = threadIdx.x & 31;
  const int ty = threadIdx.x >> 5;      // 0..7
  const int f0 = blockIdx.x * 32;
  const int d0 = blockIdx.y * 32;
#pragma unroll
  for (int i = 0; i < 4; ++i) {
    tile[ty + i * 8][tx] = wdec[(size_t)(d0 + ty + i * 8) * F_SZ + (f0 + tx)];
  }
  __syncthreads();
#pragma unroll
  for (int i = 0; i < 4; ++i) {
    wdecT[(size_t)(f0 + ty + i * 8) * D_SZ + (d0 + tx)] =
        f32_to_bf16_bits(tile[tx][ty + i * 8]);
  }
}

// ---------------------------------------------------------------------------
// Kernel 4: fused encode GEMM (bf16 WMMA, f32 accum) + b_enc + ReLU + top-64
//   Block = 32 rows of x, 16 waves. Waves 0-7 own rows 0-15, waves 8-15 own
//   rows 16-31, over the same 256-column F-tile (two 16x16 accs per wave);
//   duplicate B reads between the two M-halves hit the WGP cache, halving L2
//   traffic vs a 16-row tile. A tile (32 x 2048 bf16 = 128 KB) is staged once
//   into LDS with CDNA5 async global->LDS copies and reused for all 128
//   F-tiles. Epilogue fuses bias + ReLU + streaming per-row top-64.
// ---------------------------------------------------------------------------
__global__ __launch_bounds__(ENC_THREADS)
void k_encode_topk(const unsigned short* __restrict__ xm_,     // (B, D) bf16
                   const unsigned short* __restrict__ wencB_,  // (F, D) bf16
                   const float* __restrict__ b_enc,
                   float* __restrict__ topV_g, int* __restrict__ topI_g) {
  __shared__ __bf16 Alds[ENC_ROWS * D_SZ];        // 128 KB
  __shared__ float  acts[ENC_ROWS * ACT_PITCH];   // 33 KB
  __shared__ float  bencS[256];
  __shared__ float  topV[ENC_ROWS * K_TOP];       // 8 KB
  __shared__ int    topI[ENC_ROWS * K_TOP];       // 8 KB

  const __bf16* xm    = (const __bf16*)xm_;
  const __bf16* wencB = (const __bf16*)wencB_;

  const int tid   = threadIdx.x;
  const int lane  = tid & 31;
  const int wv    = tid >> 5;         // wave id 0..15
  const int mhalf = wv >> 3;          // 0: rows 0-15, 1: rows 16-31
  const int wcol  = wv & 7;           // column-group id 0..7
  const int khalf = lane >> 4;        // 0 or 1
  const int l16   = lane & 15;
  const int m0    = blockIdx.x * ENC_ROWS;  // first batch row of this block

  // ---- Stage A tile with async global->LDS copies (ASYNCcnt path) ----
  // 32 rows x 2048 K bf16 = 8192 x 16B chunks; per-lane LDS dest = low 32
  // bits of the generic shared address (LDS aperture truncation, ISA 10.2).
#pragma unroll 4
  for (int c = tid; c < (ENC_ROWS * D_SZ) / 8; c += ENC_THREADS) {
    const int r  = c >> 8;            // 256 chunks per row
    const int kk = (c & 255) << 3;
    const unsigned lds_off =
        (unsigned)(unsigned long long)(&Alds[r * D_SZ + kk]);
    const unsigned long long gaddr =
        (unsigned long long)(xm + (size_t)(m0 + r) * D_SZ + kk);
    asm volatile("global_load_async_to_lds_b128 %0, %1, off"
                 :: "v"(lds_off), "v"(gaddr)
                 : "memory");
  }
  for (int e = tid; e < ENC_ROWS * K_TOP; e += ENC_THREADS) {
    topV[e] = 0.0f; topI[e] = 0;
  }
  asm volatile("s_wait_asynccnt 0x0" ::: "memory");
  __syncthreads();

  // Per-scan-thread running min of its row's top-64 list (tid < 32 only).
  float cmin = 0.0f;
  int   cpos = 0;

  const int col0 = wcol * 32 + l16;   // n index of accumulator 0
  const int col1 = col0 + 16;         // n index of accumulator 1
  const int arow_base = (mhalf * 16 + l16) * D_SZ;   // A-fragment row

  for (int f0 = 0; f0 < F_SZ; f0 += 256) {
    if (tid < 256) bencS[tid] = b_enc[f0 + tid];

    v8f acc0 = {0.f, 0.f, 0.f, 0.f, 0.f, 0.f, 0.f, 0.f};
    v8f acc1 = acc0;

    // B rows (W_enc rows = output features); each lane reads 32B contiguous.
    const __bf16* brow0 = wencB + (size_t)(f0 + col0) * D_SZ + khalf * 16;
    const __bf16* brow1 = wencB + (size_t)(f0 + col1) * D_SZ + khalf * 16;

#pragma unroll 4
    for (int kb = 0; kb < D_SZ; kb += 32) {
      union { v16bf v; v8bf h[2]; } af, bfr0, bfr1;
      // A 16x32 bf16 fragment: m = l16; lanes 0-15 K=0..15,
      // lanes 16-31 K=16..31.
      const __bf16* ap = &Alds[arow_base + kb + khalf * 16];
      af.h[0] = *(const v8bf*)(ap);
      af.h[1] = *(const v8bf*)(ap + 8);
      // B 32x16 fragment: column = l16, K = khalf*16 + 0..15 (contiguous).
      bfr0.h[0] = *(const v8bf*)(brow0 + kb);
      bfr0.h[1] = *(const v8bf*)(brow0 + kb + 8);
      bfr1.h[0] = *(const v8bf*)(brow1 + kb);
      bfr1.h[1] = *(const v8bf*)(brow1 + kb + 8);

      acc0 = __builtin_amdgcn_wmma_f32_16x16x32_bf16(false, af.v, false, bfr0.v,
                                                     (short)0, acc0, false, false);
      acc1 = __builtin_amdgcn_wmma_f32_16x16x32_bf16(false, af.v, false, bfr1.v,
                                                     (short)0, acc1, false, false);
    }
    __syncthreads();  // prev scan done; bencS ready

    // Epilogue: C layout -> acts LDS with bias. m = r + 8*khalf, n = l16.
    {
      const int c0 = wcol * 32 + l16;
      const int mb = mhalf * 16;
#pragma unroll
      for (int r = 0; r < 8; ++r) {
        const int m = mb + r + 8 * khalf;
        acts[m * ACT_PITCH + c0]      = acc0[r] + bencS[c0];
        acts[m * ACT_PITCH + c0 + 16] = acc1[r] + bencS[c0 + 16];
      }
    }
    __syncthreads();

    // Streaming top-64 with ReLU (list seeded with 0 => non-positives never enter).
    if (tid < ENC_ROWS) {
      const float* arow = &acts[tid * ACT_PITCH];
      float* tv = &topV[tid * K_TOP];
      int*   ti = &topI[tid * K_TOP];
      for (int c = 0; c < 256; ++c) {
        const float v = arow[c];
        if (v > cmin) {
          tv[cpos] = v; ti[cpos] = f0 + c;
          float mn = tv[0]; int mp = 0;
#pragma unroll 8
          for (int j = 1; j < K_TOP; ++j) {
            const float t = tv[j];
            if (t < mn) { mn = t; mp = j; }
          }
          cmin = mn; cpos = mp;
        }
      }
    }
    __syncthreads();
  }

  for (int e = tid; e < ENC_ROWS * K_TOP; e += ENC_THREADS) {
    const int r = e >> 6;
    const size_t o = (size_t)(m0 + r) * K_TOP + (e & 63);
    topV_g[o] = topV[e];
    topI_g[o] = topI[e];
  }
}

// ---------------------------------------------------------------------------
// Kernel 5: sparse decode: out[b,:] = b_dec + sum_j val[b,j] * W_decT[idx[b,j],:]
//   One block per batch row; thread t owns 8 contiguous d -> 16B coalesced reads.
// ---------------------------------------------------------------------------
__global__ __launch_bounds__(256)
void k_decode(const float* __restrict__ topV_g, const int* __restrict__ topI_g,
              const unsigned short* __restrict__ wdecT_,
              const float* __restrict__ b_dec, float* __restrict__ out) {
  __shared__ float sv[K_TOP];
  __shared__ int   si[K_TOP];
  const __bf16* wdecT = (const __bf16*)wdecT_;
  const int row = blockIdx.x;
  const int tid = threadIdx.x;
  if (tid < K_TOP) {
    sv[tid] = topV_g[(size_t)row * K_TOP + tid];
    si[tid] = topI_g[(size_t)row * K_TOP + tid];
  }
  __syncthreads();

  const int d0 = tid * 8;
  float acc[8];
#pragma unroll
  for (int i = 0; i < 8; ++i) acc[i] = b_dec[d0 + i];

  for (int j = 0; j < K_TOP; ++j) {
    const float v = sv[j];           // uniform across block
    if (v != 0.0f) {
      const v8bf w = *(const v8bf*)(wdecT + (size_t)si[j] * D_SZ + d0);
#pragma unroll
      for (int i = 0; i < 8; ++i) acc[i] += v * bf16_to_f32(w[i]);
    }
  }
  float4* o = (float4*)(out + (size_t)row * D_SZ + d0);
  o[0] = make_float4(acc[0], acc[1], acc[2], acc[3]);
  o[1] = make_float4(acc[4], acc[5], acc[6], acc[7]);
}

// ---------------------------------------------------------------------------
// Host launcher
// ---------------------------------------------------------------------------
extern "C" void kernel_launch(void* const* d_in, const int* in_sizes, int n_in,
                              void* d_out, int out_size, void* d_ws, size_t ws_size,
                              hipStream_t stream) {
  (void)in_sizes; (void)n_in; (void)out_size; (void)ws_size;
  const float* x     = (const float*)d_in[0];
  const float* w_enc = (const float*)d_in[1];
  const float* b_enc = (const float*)d_in[2];
  const float* w_dec = (const float*)d_in[3];
  const float* b_dec = (const float*)d_in[4];
  // d_in[5] is k == 64 (hardcoded as K_TOP)

  char* ws = (char*)d_ws;
  size_t off = 0;
  unsigned short* xm    = (unsigned short*)(ws + off); off += (size_t)B_SZ * D_SZ * 2;  //  32 MB
  unsigned short* wencB = (unsigned short*)(ws + off); off += (size_t)F_SZ * D_SZ * 2;  // 128 MB
  unsigned short* wdecT = (unsigned short*)(ws + off); off += (size_t)F_SZ * D_SZ * 2;  // 128 MB
  float* topV = (float*)(ws + off); off += (size_t)B_SZ * K_TOP * sizeof(float);        //   2 MB
  int*   topI = (int*)(ws + off);   off += (size_t)B_SZ * K_TOP * sizeof(int);          //   2 MB

  k_prep_xm<<<B_SZ, 256, 0, stream>>>(x, b_dec, xm);

  {
    const size_t n4 = (size_t)F_SZ * D_SZ / 4;
    const int blocks = (int)((n4 + 255) / 256);
    k_f32_to_bf16_vec4<<<blocks, 256, 0, stream>>>(w_enc, wencB, n4);
  }

  k_transpose_wdec<<<dim3(F_SZ / 32, D_SZ / 32), 256, 0, stream>>>(w_dec, wdecT);

  k_encode_topk<<<B_SZ / ENC_ROWS, ENC_THREADS, 0, stream>>>(xm, wencB, b_enc,
                                                             topV, topI);

  k_decode<<<B_SZ, 256, 0, stream>>>(topV, topI, wdecT, b_dec, (float*)d_out);
}
